// PairScoreModel_89833535963206
// MI455X (gfx1250) — matrix operand order repaired and checked by reading, not compiled, not run on hardware
//
#include <hip/hip_runtime.h>
#include <hip/hip_bf16.h>

// ---- problem constants ----
#define B_  4
#define N_  4096
#define F_  512
#define HA_ 128
#define HV_ 32
#define ROWS (B_*N_)          // 16384

typedef __attribute__((ext_vector_type(16))) _Float16 v16h;
typedef __attribute__((ext_vector_type(8)))  float    v8f;
typedef __attribute__((ext_vector_type(4)))  unsigned int v4u;
typedef __attribute__((ext_vector_type(4)))  unsigned int u32x4;
typedef __attribute__((ext_vector_type(8)))  int          i32x8;
typedef __attribute__((ext_vector_type(4)))  int          i32x4;

union Frag { v16h h; v4u u[2]; };
union H8   { _Float16 h[8]; v4u u; };

// =====================================================================
// Kernel 1: pack W_act / W_v1 into f16 WMMA B-fragment layout, compute
// gen[b] = leaky_relu(gen_number*W_gen + b_gen), zero critic accumulator.
// B-frag packing: element e -> (kt, ct, lane L, kk); value =
//   W[(kt*32 + (L>=16?16:0) + kk) * COLS + ct*16 + (L&15)]
// so each lane's 16 f16 (32 B) are contiguous in the packed buffer.
// =====================================================================
__global__ void prep_pack(const float* __restrict__ gen_number,
                          const float* __restrict__ W_gen,
                          const float* __restrict__ b_gen,
                          const float* __restrict__ W_act,
                          const float* __restrict__ W_v1,
                          float* __restrict__ gen,
                          float* __restrict__ critic_acc,
                          _Float16* __restrict__ WpA,
                          _Float16* __restrict__ WpV) {
  int e = blockIdx.x * blockDim.x + threadIdx.x;
  const int NA = 16 * 8 * 512;   // 65536  (16 kt x 8 ct x 512)
  const int NV = 16 * 2 * 512;   // 16384
  if (e < NA) {
    int kt = e >> 12;
    int ct = (e >> 9) & 7;
    int L  = (e >> 4) & 31;
    int kk = e & 15;
    int col = ct * 16 + (L & 15);
    int k   = kt * 32 + ((L >> 4) << 4) + kk;
    WpA[e] = (_Float16)W_act[k * HA_ + col];
  } else if (e < NA + NV) {
    int e2 = e - NA;
    int kt = e2 >> 10;
    int ct = (e2 >> 9) & 1;
    int L  = (e2 >> 4) & 31;
    int kk = e2 & 15;
    int col = ct * 16 + (L & 15);
    int k   = kt * 32 + ((L >> 4) << 4) + kk;
    WpV[e2] = (_Float16)W_v1[k * HV_ + col];
  } else if (e < NA + NV + B_) {
    int b = e - (NA + NV);
    float x = gen_number[b] * W_gen[0] + b_gen[0];
    gen[b] = x > 0.0f ? x : 0.01f * x;   // leaky_relu, slope 0.01
    critic_acc[b] = 0.0f;
  }
}

// =====================================================================
// Kernel 2: obs f32 -> f16 (rows become direct WMMA A-fragments later)
// =====================================================================
__global__ __launch_bounds__(256) void prep_obsh(const float* __restrict__ obs,
                                                 _Float16* __restrict__ obsh) {
  size_t t = (size_t)blockIdx.x * blockDim.x + threadIdx.x;  // 1,048,576 threads
  size_t base = t * 8;
  const float4* s = (const float4*)(obs + base);
  float4 a = s[0], b = s[1];
  H8 o;
  o.h[0] = (_Float16)a.x; o.h[1] = (_Float16)a.y;
  o.h[2] = (_Float16)a.z; o.h[3] = (_Float16)a.w;
  o.h[4] = (_Float16)b.x; o.h[5] = (_Float16)b.y;
  o.h[6] = (_Float16)b.z; o.h[7] = (_Float16)b.w;
  *(v4u*)(obsh + base) = o.u;
}

// =====================================================================
// Kernel 3: h = feats@W_act + b_act (WMMA), normalize rows -> HN (f16),
// critic partial: relu(feats@W_v1 + b_v1)@W_v2 accumulated per batch.
// 1 wave per 16-row tile; 8 waves per block; 128 blocks.
// =====================================================================
__global__ __launch_bounds__(256) void stageA(
    const _Float16* __restrict__ obsh, const _Float16* __restrict__ WpA,
    const _Float16* __restrict__ WpV,  const float* __restrict__ W_act,
    const float* __restrict__ b_act,   const float* __restrict__ W_v1,
    const float* __restrict__ b_v1,    const float* __restrict__ W_v2,
    const float* __restrict__ b_v2,    const float* __restrict__ gen,
    _Float16* __restrict__ HN,         float* __restrict__ critic_acc) {
  const int lane = threadIdx.x & 31;
  const int wave = threadIdx.x >> 5;
  const int rt   = blockIdx.x * 8 + wave;   // 0..1023 row tiles
  const int r0   = rt * 16;
  const int nl   = lane & 15;
  const int half = lane >> 4;

  v8f accA[8] = {};
  v8f accV[2] = {};

  // A-fragment source row for this lane (ISA 7.12.2: A rows = lane&15)
  const _Float16* arow = obsh + ((size_t)(r0 + nl)) * F_;

  for (int kt = 0; kt < 16; ++kt) {
    Frag a;
    int koff = kt * 32 + (half ? 16 : 0);          // lanes>=16 hold K 16..31
    a.u[0] = *(const v4u*)(arow + koff);           // K koff..koff+7
    a.u[1] = *(const v4u*)(arow + koff + 8);       // K koff+8..koff+15
#pragma unroll
    for (int ct = 0; ct < 8; ++ct) {
      Frag bf;
      const _Float16* p = WpA + ((((size_t)kt * 8 + ct) * 32 + lane) << 4);
      bf.u[0] = *(const v4u*)(p);
      bf.u[1] = *(const v4u*)(p + 8);
      accA[ct] = __builtin_amdgcn_wmma_f32_16x16x32_f16(
          false, a.h, false, bf.h, (short)0, accA[ct], false, false);
    }
#pragma unroll
    for (int ct = 0; ct < 2; ++ct) {
      Frag bf;
      const _Float16* p = WpV + ((((size_t)kt * 2 + ct) * 32 + lane) << 4);
      bf.u[0] = *(const v4u*)(p);
      bf.u[1] = *(const v4u*)(p + 8);
      accV[ct] = __builtin_amdgcn_wmma_f32_16x16x32_f16(
          false, a.h, false, bf.h, (short)0, accV[ct], false, false);
    }
  }

  const int   b = r0 >> 12;        // 4096 rows per batch
  const float g = gen[b];

  // ---- actor path: add gen-row + bias, row norms, write normalized f16 ----
  float ssq[8] = {0.f, 0.f, 0.f, 0.f, 0.f, 0.f, 0.f, 0.f};
#pragma unroll
  for (int ct = 0; ct < 8; ++ct) {
    int col = ct * 16 + nl;
    float add = g * W_act[(size_t)F_ * HA_ + col] + b_act[col];
#pragma unroll
    for (int v = 0; v < 8; ++v) {
      accA[ct][v] += add;
      ssq[v] += accA[ct][v] * accA[ct][v];
    }
  }
#pragma unroll
  for (int v = 0; v < 8; ++v) {    // reduce over the 16 lanes of each half
    ssq[v] += __shfl_xor(ssq[v], 1, 32);
    ssq[v] += __shfl_xor(ssq[v], 2, 32);
    ssq[v] += __shfl_xor(ssq[v], 4, 32);
    ssq[v] += __shfl_xor(ssq[v], 8, 32);
  }
  float inv[8];
#pragma unroll
  for (int v = 0; v < 8; ++v)
    inv[v] = 1.0f / fmaxf(sqrtf(ssq[v]), 1e-8f);

#pragma unroll
  for (int ct = 0; ct < 8; ++ct) {
    int col = ct * 16 + nl;
#pragma unroll
    for (int v = 0; v < 8; ++v) {
      int row = r0 + v + half * 8;                 // C/D layout: M = v (+8 for hi half)
      HN[(size_t)row * HA_ + col] = (_Float16)(accA[ct][v] * inv[v]);
    }
  }

  // ---- critic path ----
  float svv[8] = {0.f, 0.f, 0.f, 0.f, 0.f, 0.f, 0.f, 0.f};
#pragma unroll
  for (int ct = 0; ct < 2; ++ct) {
    int col = ct * 16 + nl;
    float add = g * W_v1[(size_t)F_ * HV_ + col] + b_v1[col];
    float w2  = W_v2[col];
#pragma unroll
    for (int v = 0; v < 8; ++v) {
      float x = fmaxf(accV[ct][v] + add, 0.0f);    // relu
      svv[v] += x * w2;
    }
  }
#pragma unroll
  for (int v = 0; v < 8; ++v) {
    svv[v] += __shfl_xor(svv[v], 1, 32);
    svv[v] += __shfl_xor(svv[v], 2, 32);
    svv[v] += __shfl_xor(svv[v], 4, 32);
    svv[v] += __shfl_xor(svv[v], 8, 32);
  }
  if (nl == 0) {                   // lanes 0 and 16: 8 rows each
    float tot = 8.0f * b_v2[0];
#pragma unroll
    for (int v = 0; v < 8; ++v) tot += svv[v];
    atomicAdd(&critic_acc[b], tot);
  }
}

// =====================================================================
// TDM helper: build a D# for a 2D tile (128 rows x 256 B, f16 HN rows
// viewed as 64 dwords) and issue TENSOR_LOAD_TO_LDS.
// D# per ISA ch.8: group0 = {count|flags, lds_addr, ga_lo, ga_hi|type=2};
// group1 packs data_size=4B, tensor_dim0=64, tensor_dim1=rows,
// tile_dim0=64, tile_dim1=128, tensor_dim0_stride=64. workgroup_mask=0
// (not dispatched in a cluster). Tracked by TENSORcnt.
// This toolchain's builtin takes 6 args (g0,g1,g2,g3,extra,cpol).
// =====================================================================
#if __has_builtin(__builtin_amdgcn_tensor_load_to_lds) && \
    __has_builtin(__builtin_amdgcn_s_wait_tensorcnt)
#define USE_TDM 1
__device__ __forceinline__ void tdm_load_2d(unsigned lds_off, const void* gptr,
                                            unsigned rows) {
  unsigned long long ga = (unsigned long long)(uintptr_t)gptr;
  u32x4 g0;
  g0[0] = 1u;                                            // count=1, user desc
  g0[1] = lds_off;                                       // lds_addr (bytes)
  g0[2] = (unsigned)(ga & 0xffffffffu);                  // global_addr[31:0]
  g0[3] = (unsigned)((ga >> 32) & 0x01ffffffu)           // global_addr[56:32]
          | (2u << 30);                                  // type = 2 ("image")
  i32x8 g1;
  g1[0] = 0x00020000;              // workgroup_mask=0, data_size=2 (4 bytes)
  g1[1] = (int)(64u << 16);        // tensor_dim0 = 64 dwords (low 16 bits)
  g1[2] = (int)(rows << 16);       // tensor_dim0 hi=0 | tensor_dim1 low16
  g1[3] = (int)(64u << 16);        // tensor_dim1 hi=0 | tile_dim0 = 64
  g1[4] = 128;                     // tile_dim1 = 128 rows, tile_dim2 = 0
  g1[5] = 64;                      // tensor_dim0_stride = 64 dwords (low 32)
  g1[6] = 0;                       // stride0 hi | tensor_dim1_stride low16
  g1[7] = 0;                       // tensor_dim1_stride hi
  i32x4 z4 = {0, 0, 0, 0};         // groups 2/3 unused (2D tensor)
  i32x8 z8 = {0, 0, 0, 0, 0, 0, 0, 0};
  __builtin_amdgcn_tensor_load_to_lds(g0, g1, z4, z4, z8, 0);
}
#else
#define USE_TDM 0
#endif

// =====================================================================
// Kernel 4: actor = -(HN @ HN^T) per batch. 128x128 tile per block,
// HN row/col blocks DMA'd into LDS by the Tensor Data Mover,
// 8 waves x (2x4 16x16 subtiles) of v_wmma_f32_16x16x32_f16.
// =====================================================================
__global__ __launch_bounds__(256) void stageB(const _Float16* __restrict__ HN,
                                              float* __restrict__ out) {
  __shared__ _Float16 Ar[128 * HA_];   // 32 KB: output-row block
  __shared__ _Float16 Bc[128 * HA_];   // 32 KB: output-col block
  const int bz = blockIdx.z;
  const int n0 = blockIdx.y * 128;
  const int m0 = blockIdx.x * 128;
  const _Float16* hb = HN + (size_t)bz * N_ * HA_;

#if USE_TDM
  if (threadIdx.x < 32) {   // wave 0 issues both DMAs (one instr per wave)
    tdm_load_2d((unsigned)(uintptr_t)(void*)Ar, hb + (size_t)n0 * HA_, N_);
    tdm_load_2d((unsigned)(uintptr_t)(void*)Bc, hb + (size_t)m0 * HA_, N_);
    __builtin_amdgcn_s_wait_tensorcnt(0);   // TDM wrote LDS
  }
  __syncthreads();                          // publish to waves 1..7
#else
  {  // fallback: flat 32KB copies (rows are contiguous 128 f16)
    const v4u* sA = (const v4u*)(hb + (size_t)n0 * HA_);
    const v4u* sB = (const v4u*)(hb + (size_t)m0 * HA_);
    v4u* dA = (v4u*)Ar;
    v4u* dB = (v4u*)Bc;
    for (int c = threadIdx.x; c < 2048; c += 256) { dA[c] = sA[c]; dB[c] = sB[c]; }
  }
  __syncthreads();
#endif

  const int lane = threadIdx.x & 31;
  const int wave = threadIdx.x >> 5;
  const int wrow = wave >> 1;        // 0..3 -> 32-row strip
  const int wcol = wave & 1;         // 0..1 -> 64-col strip
  const int nl   = lane & 15;
  const int half = lane >> 4;

  v8f acc[2][4] = {};

  for (int kt = 0; kt < 4; ++kt) {
    Frag a[2];
#pragma unroll
    for (int sr = 0; sr < 2; ++sr) {
      int rl = wrow * 32 + sr * 16 + nl;
      int ko = kt * 32 + (half ? 16 : 0);
      a[sr].u[0] = *(const v4u*)(Ar + rl * HA_ + ko);
      a[sr].u[1] = *(const v4u*)(Ar + rl * HA_ + ko + 8);
    }
#pragma unroll
    for (int sc = 0; sc < 4; ++sc) {
      Frag bf;
      int ml = wcol * 64 + sc * 16 + nl;
      int kb = kt * 32 + (half ? 16 : 0);
      bf.u[0] = *(const v4u*)(Bc + ml * HA_ + kb);
      bf.u[1] = *(const v4u*)(Bc + ml * HA_ + kb + 8);
#pragma unroll
      for (int sr = 0; sr < 2; ++sr)
        acc[sr][sc] = __builtin_amdgcn_wmma_f32_16x16x32_f16(
            false, a[sr].h, false, bf.h, (short)0, acc[sr][sc], false, false);
    }
  }

  float* ob = out + (size_t)bz * ((size_t)N_ * N_);
#pragma unroll
  for (int sr = 0; sr < 2; ++sr)
#pragma unroll
    for (int sc = 0; sc < 4; ++sc) {
      int colg = m0 + wcol * 64 + sc * 16 + nl;
#pragma unroll
      for (int v = 0; v < 8; ++v) {
        int rowg = n0 + wrow * 32 + sr * 16 + v + half * 8;
        ob[(size_t)rowg * N_ + colg] = -acc[sr][sc][v];
      }
    }
}

// =====================================================================
// Kernel 5: critic mean
// =====================================================================
__global__ void finalize(const float* __restrict__ critic_acc,
                         float* __restrict__ out) {
  int b = threadIdx.x;
  if (b < B_) out[(size_t)B_ * N_ * N_ + b] = critic_acc[b] * (1.0f / (float)N_);
}

// =====================================================================
extern "C" void kernel_launch(void* const* d_in, const int* in_sizes, int n_in,
                              void* d_out, int out_size, void* d_ws, size_t ws_size,
                              hipStream_t stream) {
  (void)in_sizes; (void)n_in; (void)out_size; (void)ws_size;
  const float* obs        = (const float*)d_in[0];
  const float* gen_number = (const float*)d_in[1];
  const float* W_gen      = (const float*)d_in[2];
  const float* b_gen      = (const float*)d_in[3];
  const float* W_act      = (const float*)d_in[4];
  const float* b_act      = (const float*)d_in[5];
  const float* W_v1       = (const float*)d_in[6];
  const float* b_v1       = (const float*)d_in[7];
  const float* W_v2       = (const float*)d_in[8];
  const float* b_v2       = (const float*)d_in[9];
  float* out = (float*)d_out;

  // workspace layout (~21.1 MB)
  char* ws = (char*)d_ws;
  float*    gen        = (float*)(ws + 0);
  float*    critic_acc = (float*)(ws + 64);
  _Float16* WpA        = (_Float16*)(ws + 256);                       // 128 KB
  _Float16* WpV        = (_Float16*)(ws + 256 + 131072);              //  32 KB
  _Float16* obsh       = (_Float16*)(ws + 256 + 131072 + 32768);      //  16 MB
  _Float16* HN         = (_Float16*)(ws + 256 + 131072 + 32768 + 16777216); // 4 MB

  prep_pack<<<321, 256, 0, stream>>>(gen_number, W_gen, b_gen, W_act, W_v1,
                                     gen, critic_acc, WpA, WpV);
  prep_obsh<<<4096, 256, 0, stream>>>(obs, obsh);
  stageA<<<128, 256, 0, stream>>>(obsh, WpA, WpV, W_act, b_act, W_v1, b_v1,
                                  W_v2, b_v2, gen, HN, critic_acc);
  stageB<<<dim3(32, 32, B_), 256, 0, stream>>>(HN, out);
  finalize<<<1, 64, 0, stream>>>(critic_acc, out);
}
